// BMoE_57767310131676
// MI455X (gfx1250) — matrix operand order repaired and checked by reading, not compiled, not run on hardware
//
#include <hip/hip_runtime.h>

typedef __attribute__((ext_vector_type(8)))  float  v8f;
typedef __attribute__((ext_vector_type(4)))  __bf16 v4bf;
typedef __attribute__((ext_vector_type(8)))  __bf16 v8bf;
typedef __attribute__((ext_vector_type(16))) __bf16 v16bf;

#define NB    65536
#define DIN   512
#define NE    8
#define DH    128
#define DOUT  128
#define HS    136   // padded LDS row stride in halves (272B -> rotates banks by 4/row)
#define NWAVE 4     // waves per block
#define MROWS 32    // rows per wave (two 16-row M-tiles)

#define NW_HALVES (NE * DH * DIN + 3 * NE * DH * DH)   // 917504 weight halves

__device__ __forceinline__ __bf16 f2bf(float f) {
    return static_cast<__bf16>(f);   // hardware f32->bf16 (RNE) on gfx1250
}

__device__ __forceinline__ v16bf cat16(v8bf lo, v8bf hi) {
    return __builtin_shufflevector(lo, hi, 0,1,2,3,4,5,6,7,8,9,10,11,12,13,14,15);
}

// ---------------------------------------------------------------------------
// Prep A: W0..W3 fp32 (E,K,N) -> bf16 transposed [e][n][k] in workspace.
// ---------------------------------------------------------------------------
__global__ void moe_prep_w(const float* __restrict__ W0, const float* __restrict__ W1,
                           const float* __restrict__ W2, const float* __restrict__ W3,
                           __bf16* __restrict__ wt) {
    int tid = blockIdx.x * 256 + threadIdx.x;
    const int n0 = NE * DH * DIN;   // 524288
    const int n1 = NE * DH * DH;    // 131072
    if (tid < n0) {
        int e = tid / (DH * DIN);
        int r = tid % (DH * DIN);
        int n = r / DIN, k = r % DIN;
        wt[tid] = f2bf(W0[(e * DIN + k) * DH + n]);
    } else if (tid < n0 + 3 * n1) {
        int t = tid - n0;
        int layer = t / n1;
        int u = t % n1;
        int e = u / (DH * DH);
        int r = u % (DH * DH);
        int n = r / DH, k = r % DH;
        const float* W = (layer == 0) ? W1 : (layer == 1) ? W2 : W3;
        wt[tid] = f2bf(W[(e * DH + k) * DH + n]);
    }
}

// ---------------------------------------------------------------------------
// Prep B: x fp32 -> bf16 (row-major copy), 4 elements / thread.
// ---------------------------------------------------------------------------
__global__ void moe_prep_x(const float* __restrict__ x, __bf16* __restrict__ xb) {
    long t = ((long)blockIdx.x * 256 + threadIdx.x) * 4;
    float4 v = *(const float4*)(x + t);
    v4bf o;
    o[0] = f2bf(v.x); o[1] = f2bf(v.y); o[2] = f2bf(v.z); o[3] = f2bf(v.w);
    *(v4bf*)(xb + t) = o;
}

// ---------------------------------------------------------------------------
// Fused MoE: gate softmax + 4-layer expert stack + alpha-weighted reduction.
// 4 independent waves / block; each wave owns 32 rows = two 16-row M-tiles,
// so every weight (B) operand load feeds two WMMAs.
// XB: layer-0 A operand from pre-converted bf16 x (workspace) vs fp32 x.
// ---------------------------------------------------------------------------
template <bool XB>
__launch_bounds__(128)
__global__ void moe_fused(const float* __restrict__ x,
                          const __bf16* __restrict__ xb,
                          const float* __restrict__ Wg,
                          const float* __restrict__ bg,
                          const __bf16* __restrict__ wt,
                          float* __restrict__ out) {
    __shared__ __bf16 hbuf[NWAVE][2][16 * HS];  // per-wave, per-tile activations
    __shared__ float  gbuf[NWAVE][MROWS * NE];  // per-wave gate logits -> alpha

    const int lane = threadIdx.x & 31;
    const int wv   = threadIdx.x >> 5;
    const int hi   = lane >> 4;                 // lane half (A/B K-interleave select)
    const int ln   = lane & 15;
    const long rowBase = (long)blockIdx.x * (NWAVE * MROWS) + (long)wv * MROWS;

    const __bf16* w0t = wt;
    const __bf16* w1t = wt + (long)NE * DH * DIN;
    const __bf16* w2t = w1t + (long)NE * DH * DH;
    const __bf16* w3t = w2t + (long)NE * DH * DH;

    // ---------------- gate: logits + softmax -> alpha in gbuf ----------------
    {
        const int e  = lane & 7;
        const int rq = lane >> 3;               // 0..3 : rows rq*8 .. rq*8+7
        float ga[8];
        #pragma unroll
        for (int r = 0; r < 8; ++r) ga[r] = bg[e];
        const float* xg = x + (rowBase + rq * 8) * DIN;
        for (int k = 0; k < DIN; k += 4) {
            float w0 = Wg[(k + 0) * NE + e];
            float w1 = Wg[(k + 1) * NE + e];
            float w2 = Wg[(k + 2) * NE + e];
            float w3 = Wg[(k + 3) * NE + e];
            #pragma unroll
            for (int r = 0; r < 8; ++r) {
                float4 a = *(const float4*)(xg + (long)r * DIN + k);
                ga[r] += a.x * w0 + a.y * w1 + a.z * w2 + a.w * w3;
            }
        }
        #pragma unroll
        for (int r = 0; r < 8; ++r) gbuf[wv][(rq * 8 + r) * NE + e] = ga[r];
        asm volatile("s_wait_dscnt 0x0" ::: "memory");
        {
            float l[NE];
            float m = -3.0e38f;
            #pragma unroll
            for (int j = 0; j < NE; ++j) { l[j] = gbuf[wv][lane * NE + j]; m = fmaxf(m, l[j]); }
            float s = 0.f;
            #pragma unroll
            for (int j = 0; j < NE; ++j) { l[j] = __expf(l[j] - m); s += l[j]; }
            float inv = 1.0f / s;
            #pragma unroll
            for (int j = 0; j < NE; ++j) gbuf[wv][lane * NE + j] = l[j] * inv;
        }
        asm volatile("s_wait_dscnt 0x0" ::: "memory");
    }

    const v8f vzero = {0.f, 0.f, 0.f, 0.f, 0.f, 0.f, 0.f, 0.f};
    v8f outAcc0[8], outAcc1[8];
    #pragma unroll
    for (int i = 0; i < 8; ++i) { outAcc0[i] = vzero; outAcc1[i] = vzero; }

    const float*  xrf0 = x  + (rowBase + ln) * DIN;        // tile-0 fp32 A rows
    const float*  xrf1 = xrf0 + 16 * DIN;                  // tile-1
    const __bf16* xrb0 = xb + (rowBase + ln) * DIN;        // tile-0 bf16 A rows
    const __bf16* xrb1 = xrb0 + 16 * DIN;                  // tile-1

    #pragma unroll 1
    for (int e = 0; e < NE; ++e) {
        v8f acc0[8], acc1[8];
        #pragma unroll
        for (int i = 0; i < 8; ++i) { acc0[i] = vzero; acc1[i] = vzero; }

        // ----- layer 0: two (16x512)@W0t[e] tiles sharing every B operand -----
        const __bf16* w0e = w0t + (long)e * DH * DIN;
        for (int kb = 0; kb < DIN / 32; ++kb) {
            // A layout (16-bit, 16x32): lanes<16 hold K {0..7,16..23}, lanes>=16 {8..15,24..31}
            v16bf A0, A1;
            if (XB) {
                const __bf16* a0p = xrb0 + kb * 32 + hi * 8;
                const __bf16* a1p = xrb1 + kb * 32 + hi * 8;
                A0 = cat16(*(const v8bf*)a0p, *(const v8bf*)(a0p + 16));
                A1 = cat16(*(const v8bf*)a1p, *(const v8bf*)(a1p + 16));
            } else {
                #pragma unroll
                for (int t = 0; t < 2; ++t) {
                    const float* ap = (t ? xrf1 : xrf0) + kb * 32 + hi * 8;
                    float4 a0 = *(const float4*)(ap);
                    float4 a1 = *(const float4*)(ap + 4);
                    float4 a2 = *(const float4*)(ap + 16);
                    float4 a3 = *(const float4*)(ap + 20);
                    v16bf A;
                    A[0]  = f2bf(a0.x); A[1]  = f2bf(a0.y); A[2]  = f2bf(a0.z); A[3]  = f2bf(a0.w);
                    A[4]  = f2bf(a1.x); A[5]  = f2bf(a1.y); A[6]  = f2bf(a1.z); A[7]  = f2bf(a1.w);
                    A[8]  = f2bf(a2.x); A[9]  = f2bf(a2.y); A[10] = f2bf(a2.z); A[11] = f2bf(a2.w);
                    A[12] = f2bf(a3.x); A[13] = f2bf(a3.y); A[14] = f2bf(a3.z); A[15] = f2bf(a3.w);
                    if (t) A1 = A; else A0 = A;
                }
            }
            #pragma unroll
            for (int nt = 0; nt < 8; ++nt) {
                // B layout (32x16): lanes<16 col=ln K {0..7,16..23}, lanes>=16 col=ln K {8..15,24..31}
                const __bf16* bp = w0e + (long)(nt * 16 + ln) * DIN + kb * 32 + hi * 8;
                v16bf Bm = cat16(*(const v8bf*)bp, *(const v8bf*)(bp + 16));
                acc0[nt] = __builtin_amdgcn_wmma_f32_16x16x32_bf16(
                    false, A0, false, Bm, (short)0, acc0[nt], false, false);
                acc1[nt] = __builtin_amdgcn_wmma_f32_16x16x32_bf16(
                    false, A1, false, Bm, (short)0, acc1[nt], false, false);
            }
        }

        // relu -> bf16 -> hbuf  (D layout: lane ln = col, vgpr v -> row v+8*hi)
        #pragma unroll
        for (int nt = 0; nt < 8; ++nt) {
            #pragma unroll
            for (int v = 0; v < 8; ++v) {
                hbuf[wv][0][(v + 8 * hi) * HS + nt * 16 + ln] = f2bf(fmaxf(acc0[nt][v], 0.f));
                hbuf[wv][1][(v + 8 * hi) * HS + nt * 16 + ln] = f2bf(fmaxf(acc1[nt][v], 0.f));
            }
        }
        asm volatile("s_wait_dscnt 0x0" ::: "memory");

        // ----- layers 1,2: (16x128)@(128x128) x2 tiles, A from LDS, relu back -----
        #pragma unroll 1
        for (int L = 0; L < 2; ++L) {
            const __bf16* wle = ((L == 0) ? w1t : w2t) + (long)e * DH * DH;
            #pragma unroll
            for (int i = 0; i < 8; ++i) { acc0[i] = vzero; acc1[i] = vzero; }
            #pragma unroll
            for (int kb = 0; kb < DH / 32; ++kb) {
                const __bf16* a0p = &hbuf[wv][0][ln * HS + kb * 32 + hi * 8];
                const __bf16* a1p = &hbuf[wv][1][ln * HS + kb * 32 + hi * 8];
                v16bf A0 = cat16(*(const v8bf*)a0p, *(const v8bf*)(a0p + 16));
                v16bf A1 = cat16(*(const v8bf*)a1p, *(const v8bf*)(a1p + 16));
                #pragma unroll
                for (int nt = 0; nt < 8; ++nt) {
                    const __bf16* bp = wle + (long)(nt * 16 + ln) * DH + kb * 32 + hi * 8;
                    v16bf Bm = cat16(*(const v8bf*)bp, *(const v8bf*)(bp + 16));
                    acc0[nt] = __builtin_amdgcn_wmma_f32_16x16x32_bf16(
                        false, A0, false, Bm, (short)0, acc0[nt], false, false);
                    acc1[nt] = __builtin_amdgcn_wmma_f32_16x16x32_bf16(
                        false, A1, false, Bm, (short)0, acc1[nt], false, false);
                }
            }
            #pragma unroll
            for (int nt = 0; nt < 8; ++nt) {
                #pragma unroll
                for (int v = 0; v < 8; ++v) {
                    hbuf[wv][0][(v + 8 * hi) * HS + nt * 16 + ln] = f2bf(fmaxf(acc0[nt][v], 0.f));
                    hbuf[wv][1][(v + 8 * hi) * HS + nt * 16 + ln] = f2bf(fmaxf(acc1[nt][v], 0.f));
                }
            }
            asm volatile("s_wait_dscnt 0x0" ::: "memory");
        }

        // ----- layer 3 (no relu) + alpha-weighted accumulation -----
        {
            const __bf16* w3e = w3t + (long)e * DH * DH;
            #pragma unroll
            for (int i = 0; i < 8; ++i) { acc0[i] = vzero; acc1[i] = vzero; }
            #pragma unroll
            for (int kb = 0; kb < DH / 32; ++kb) {
                const __bf16* a0p = &hbuf[wv][0][ln * HS + kb * 32 + hi * 8];
                const __bf16* a1p = &hbuf[wv][1][ln * HS + kb * 32 + hi * 8];
                v16bf A0 = cat16(*(const v8bf*)a0p, *(const v8bf*)(a0p + 16));
                v16bf A1 = cat16(*(const v8bf*)a1p, *(const v8bf*)(a1p + 16));
                #pragma unroll
                for (int nt = 0; nt < 8; ++nt) {
                    const __bf16* bp = w3e + (long)(nt * 16 + ln) * DH + kb * 32 + hi * 8;
                    v16bf Bm = cat16(*(const v8bf*)bp, *(const v8bf*)(bp + 16));
                    acc0[nt] = __builtin_amdgcn_wmma_f32_16x16x32_bf16(
                        false, A0, false, Bm, (short)0, acc0[nt], false, false);
                    acc1[nt] = __builtin_amdgcn_wmma_f32_16x16x32_bf16(
                        false, A1, false, Bm, (short)0, acc1[nt], false, false);
                }
            }
            float av0[8], av1[8];
            #pragma unroll
            for (int v = 0; v < 8; ++v) {
                av0[v] = gbuf[wv][(v + 8 * hi) * NE + e];
                av1[v] = gbuf[wv][(16 + v + 8 * hi) * NE + e];
            }
            #pragma unroll
            for (int nt = 0; nt < 8; ++nt) {
                #pragma unroll
                for (int v = 0; v < 8; ++v) {
                    outAcc0[nt][v] += av0[v] * acc0[nt][v];
                    outAcc1[nt][v] += av1[v] * acc1[nt][v];
                }
            }
        }
    }

    // ----- store output (fp32), coalesced 16-lane rows -----
    #pragma unroll
    for (int nt = 0; nt < 8; ++nt) {
        #pragma unroll
        for (int v = 0; v < 8; ++v) {
            out[(rowBase + v + 8 * hi) * DOUT + nt * 16 + ln]      = outAcc0[nt][v];
            out[(rowBase + 16 + v + 8 * hi) * DOUT + nt * 16 + ln] = outAcc1[nt][v];
        }
    }
}

// ---------------------------------------------------------------------------
extern "C" void kernel_launch(void* const* d_in, const int* in_sizes, int n_in,
                              void* d_out, int out_size, void* d_ws, size_t ws_size,
                              hipStream_t stream) {
    const float* x  = (const float*)d_in[0];
    const float* Wg = (const float*)d_in[1];
    const float* bg = (const float*)d_in[2];
    const float* W0 = (const float*)d_in[3];
    const float* W1 = (const float*)d_in[4];
    const float* W2 = (const float*)d_in[5];
    const float* W3 = (const float*)d_in[6];
    __bf16* wt = (__bf16*)d_ws;                        // weights: 917504 halves (~1.75 MB)
    __bf16* xb = wt + NW_HALVES;                       // x_bf16:  33.5M halves (64 MB)
    float* out = (float*)d_out;

    const size_t needXB = (size_t)NW_HALVES * 2 + (size_t)NB * DIN * 2;
    const bool   useXB  = ws_size >= needXB;

    moe_prep_w<<<dim3((NW_HALVES + 255) / 256), dim3(256), 0, stream>>>(W0, W1, W2, W3, wt);
    const unsigned grid = NB / (NWAVE * MROWS);        // 512 blocks of 128 threads
    if (useXB) {
        const long xquads = (long)NB * DIN / 4;        // 8,388,608 threads
        moe_prep_x<<<dim3((unsigned)(xquads / 256)), dim3(256), 0, stream>>>(x, xb);
        moe_fused<true><<<dim3(grid), dim3(128), 0, stream>>>(x, xb, Wg, bg, wt, out);
    } else {
        moe_fused<false><<<dim3(grid), dim3(128), 0, stream>>>(x, xb, Wg, bg, wt, out);
    }
}
